// Net_49581102465490
// MI455X (gfx1250) — matrix operand order repaired and checked by reading, not compiled, not run on hardware
//
#include <hip/hip_runtime.h>

typedef __attribute__((ext_vector_type(16))) _Float16 v16h;
typedef __attribute__((ext_vector_type(8)))  _Float16 v8h;
typedef __attribute__((ext_vector_type(8)))  float    v8f;

#define WFENCE() asm volatile("" ::: "memory")

constexpr int H    = 40;
constexpr int G3   = 120;   // 3*H
constexpr int Bsz  = 4096;
constexpr int LIN  = 1024;
constexpr int LOUT = 128;
constexpr int WAVES = 2;            // waves per workgroup
constexpr int ROWS  = WAVES * 16;   // batch rows per workgroup (16 per wave = WMMA N)
constexpr int CH    = 32;           // encoder input chunk (timesteps) staged in LDS

// ---- LDS layout (bytes); single __shared__ block -> LDS base offset 0 ----
constexpr int OFF_A0   = 0;                                   // 128x64 f16 : W_hh (enc, then dec)
constexpr int OFF_A1   = OFF_A0 + 128*64*2;                   // 128x64 f16 : W_ih (dec)
constexpr int OFF_GH   = OFF_A1 + 128*64*2;                   // WAVES*16*128 f32 : gh scratch
constexpr int OFF_GI   = OFF_GH + WAVES*16*128*4;             // WAVES*16*128 f32 : dec gi / enc x-chunk (union)
constexpr int OFF_H16  = OFF_GI + WAVES*16*128*4;             // WAVES*16*64 f16  : h (WMMA B operand, K-padded)
constexpr int OFF_HF   = OFF_H16 + WAVES*16*64*2;             // WAVES*16*40 f32  : h (exact)
constexpr int OFF_DECX = OFF_HF + WAVES*16*40*4;              // WAVES*16*64 f16  : dec_x (B operand)
constexpr int OFF_VEC  = OFF_DECX + WAVES*16*64*2;            // small packed vectors
constexpr int LDS_BYTES = OFF_VEC + 4096;                     // ~81 KB

__device__ __forceinline__ float sigm(float x) { return 1.f / (1.f + __expf(-x)); }
__device__ __forceinline__ float tanh_(float x) {
  float t = fminf(fmaxf(-2.f * x, -30.f), 30.f);
  float e = __expf(t);
  return (1.f - e) / (1.f + e);
}

// A operand tile (16x32 f16), A stored row-major [128][64] in LDS, zero-padded.
// contiguous k order used here (same as ldsB): lanes 0-15 row M=lane, K {0..15}; lanes 16-31 row M=lane-16, K {16..31}.
__device__ __forceinline__ v16h ldsA(const _Float16* A, int m, int k, int lane) {
  const _Float16* p = A + ((m << 4) + (lane & 15)) * 64 + (k << 5) + ((lane >> 4) << 4);
  return *(const v16h*)p;
}
// B operand tile (32x16 f16) = h^T. Lane n holds column N=n&15, K-range (n>>4)*16 .. +16 (+ kstep*32).
__device__ __forceinline__ v16h ldsB(const _Float16* Hb, int k, int lane) {
  return *(const v16h*)(Hb + (lane & 15) * 64 + (k << 5) + ((lane >> 4) << 4));
}

// D = A(LDS)[128x64] x B[64x16] -> OUT[16 cols][128 g] f32 in LDS
__device__ __forceinline__ void gemm_ldsA(const _Float16* A, const _Float16* Bb, float* OUT, int lane) {
  v16h b0 = ldsB(Bb, 0, lane);
  v16h b1 = ldsB(Bb, 1, lane);
  const int col = lane & 15;
  const int half8 = (lane >> 4) << 3;
#pragma unroll
  for (int m = 0; m < 8; ++m) {
    v8f acc = {};
    acc = __builtin_amdgcn_wmma_f32_16x16x32_f16(false, ldsA(A, m, 0, lane), false, b0,
                                                 (short)0, acc, false, false);
    acc = __builtin_amdgcn_wmma_f32_16x16x32_f16(false, ldsA(A, m, 1, lane), false, b1,
                                                 (short)0, acc, false, false);
    *(v8f*)(OUT + col * 128 + (m << 4) + half8) = acc;
  }
}

// D = A(regs) x B -> OUT ; A held in 16 v16h registers (loop-invariant W_hh for the encoder)
__device__ __forceinline__ void gemm_regA(const v16h* Ak0, const v16h* Ak1,
                                          const _Float16* Bb, float* OUT, int lane) {
  v16h b0 = ldsB(Bb, 0, lane);
  v16h b1 = ldsB(Bb, 1, lane);
  const int col = lane & 15;
  const int half8 = (lane >> 4) << 3;
#pragma unroll
  for (int m = 0; m < 8; ++m) {
    v8f acc = {};
    acc = __builtin_amdgcn_wmma_f32_16x16x32_f16(false, Ak0[m], false, b0, (short)0, acc, false, false);
    acc = __builtin_amdgcn_wmma_f32_16x16x32_f16(false, Ak1[m], false, b1, (short)0, acc, false, false);
    *(v8f*)(OUT + col * 128 + (m << 4) + half8) = acc;
  }
}

// fill [128][64] f16 LDS tile from row-major [120][40] f32 weights, zero pad
__device__ __forceinline__ void loadA(_Float16* dst, const float* __restrict__ src, int tid) {
  for (int i = tid; i < 128 * 64; i += 64) {
    int m = i >> 6, k = i & 63;
    float v = (m < G3 && k < H) ? src[m * H + k] : 0.f;
    dst[i] = (_Float16)v;
  }
}

__global__ __launch_bounds__(64) void seq2seq_gru(
    const float* __restrict__ in,   const float* __restrict__ tgt,
    const float* __restrict__ eew,  const float* __restrict__ eeb,
    const float* __restrict__ elg,  const float* __restrict__ elb,
    const float* __restrict__ ewih, const float* __restrict__ ewhh,
    const float* __restrict__ ebih, const float* __restrict__ ebhh,
    const float* __restrict__ dew,  const float* __restrict__ deb,
    const float* __restrict__ dlg,  const float* __restrict__ dlb,
    const float* __restrict__ dwih, const float* __restrict__ dwhh,
    const float* __restrict__ dbih, const float* __restrict__ dbhh,
    const float* __restrict__ ow,   const float* __restrict__ ob,
    float* __restrict__ out)
{
  __shared__ __align__(64) unsigned char smem[LDS_BYTES];
  const int tid  = threadIdx.x;
  const int lane = tid & 31;
  const int wave = tid >> 5;

  _Float16* A0  = (_Float16*)(smem + OFF_A0);
  _Float16* A1  = (_Float16*)(smem + OFF_A1);
  float*    GHp = (float*)(smem + OFF_GH) + wave * 16 * 128;
  float*    GIp = (float*)(smem + OFF_GI) + wave * 16 * 128;
  float*    XCH = (float*)(smem + OFF_GI) + wave * 16 * CH;   // union: encoder only
  _Float16* H16 = (_Float16*)(smem + OFF_H16) + wave * 16 * 64;
  float*    HF  = (float*)(smem + OFF_HF) + wave * 16 * 40;
  _Float16* DX  = (_Float16*)(smem + OFF_DECX) + wave * 16 * 64;
  // packed small vectors: E4[g]={U,V,C0,enc_b_hh}; D2[g]={dec_b_ih,dec_b_hh}
  float4* E4  = (float4*)(smem + OFF_VEC);               // 120 * 16B
  float2* D2  = (float2*)(smem + OFF_VEC + 1920);        // 120 * 8B
  float*  DWv = (float*)(smem + OFF_VEC + 2880);         // 40
  float*  DBv = DWv + 40;
  float*  LNGv = DWv + 80;
  float*  LNBv = DWv + 120;
  float*  OWv  = DWv + 160;

  const int rowBase = blockIdx.x * ROWS + wave * 16;

  // ---------------- init ----------------
  loadA(A0, ewhh, tid);  // encoder W_hh
  {
    _Float16* h16a = (_Float16*)(smem + OFF_H16);
    _Float16* dxa  = (_Float16*)(smem + OFF_DECX);
    for (int i = tid; i < WAVES * 16 * 64; i += 64) { h16a[i] = (_Float16)0.f; dxa[i] = (_Float16)0.f; }
    float* hfa = (float*)(smem + OFF_HF);
    for (int i = tid; i < WAVES * 16 * 40; i += 64) hfa[i] = 0.f;
  }
  // encoder emb+LayerNorm collapse: enc_x = s(x)*x*(wc*g) + s(x)*(bc*g) + lb
  float mw = 0.f, mb = 0.f;
  for (int j = 0; j < H; ++j) { mw += eew[j]; mb += eeb[j]; }
  mw *= (1.f / H); mb *= (1.f / H);
  float Ac = 0.f, Cc = 0.f, Dc = 0.f;
  for (int j = 0; j < H; ++j) {
    float wc = eew[j] - mw, bc = eeb[j] - mb;
    Ac += wc * wc; Cc += wc * bc; Dc += bc * bc;
  }
  Ac *= (1.f / H); Cc *= (1.f / H); Dc *= (1.f / H);
  for (int g = tid; g < G3; g += 64) {
    float u = 0.f, v = 0.f, c0 = 0.f;
    for (int j = 0; j < H; ++j) {
      float wij = ewih[g * H + j];
      float wc = eew[j] - mw, bc = eeb[j] - mb;
      u  += wij * wc * elg[j];
      v  += wij * bc * elg[j];
      c0 += wij * elb[j];
    }
    E4[g] = make_float4(u, v, c0 + ebih[g], ebhh[g]);
    D2[g] = make_float2(dbih[g], dbhh[g]);
  }
  for (int j = tid; j < H; j += 64) {
    DWv[j] = dew[j]; DBv[j] = deb[j]; LNGv[j] = dlg[j]; LNBv[j] = dlb[j]; OWv[j] = ow[j];
  }
  const float outb = ob[0];
  __syncthreads();

  const int rr = lane & 15;          // batch row within wave tile
  const int j0 = (lane >> 4) * 20;   // hidden-dim split across half-waves
  const int gbrow = rowBase + rr;

  // hoist encoder W_hh WMMA tiles into registers (loop-invariant for 1024 steps)
  v16h Ak0[8], Ak1[8];
#pragma unroll
  for (int m = 0; m < 8; ++m) { Ak0[m] = ldsA(A0, m, 0, lane); Ak1[m] = ldsA(A0, m, 1, lane); }

  // per-lane async-copy addresses for input staging (16 rows x CH floats per wave)
  const int xr = lane >> 1;              // row 0..15
  const int xc = (lane & 1) * 16;        // float column 0 / 16
  const unsigned xlds = (unsigned)(OFF_GI + (wave * 16 * CH + xr * CH + xc) * 4);

  // ---------------- encoder scan (1024 steps) ----------------
  for (int t = 0; t < LIN; ++t) {
    if ((t & (CH - 1)) == 0) {
      // async global->LDS staging of the next CH timesteps (64B per lane), ASYNCcnt-tracked
      const float* src = in + (size_t)(rowBase + xr) * LIN + t + xc;
      unsigned long long ga = (unsigned long long)(uintptr_t)src;
      asm volatile(
          "global_load_async_to_lds_b128 %0, %1, off\n\t"
          "global_load_async_to_lds_b128 %0, %1, off offset:16\n\t"
          "global_load_async_to_lds_b128 %0, %1, off offset:32\n\t"
          "global_load_async_to_lds_b128 %0, %1, off offset:48"
          :: "v"(xlds), "v"(ga) : "memory");
      if (t + CH < LIN) __builtin_prefetch(src + CH, 0, 1);   // global_prefetch_b8 for next chunk
    }
    // gh = h @ W_hh^T  (16x v_wmma_f32_16x16x32_f16, A from registers) — overlaps async copy
    gemm_regA(Ak0, Ak1, H16, GHp, lane);
    WFENCE();
    asm volatile("s_wait_asynccnt 0x0" ::: "memory");   // staged x visible before first read
    // gates (per-lane: one batch row, 20 hidden dims)
    float x = XCH[rr * CH + (t & (CH - 1))];
    float q = fmaf(x * x, Ac, fmaf(2.f * x, Cc, Dc)) + 1e-5f;
    float s = rsqrtf(q);
    float al = s * x, be = s;
    const float* gh = GHp + rr * 128;
    float* hf = HF + rr * 40;
    _Float16* h16 = H16 + rr * 64;
#pragma unroll 4
    for (int jj = 0; jj < 20; ++jj) {
      int j = j0 + jj;
      float4 er = E4[j], ez = E4[40 + j], en = E4[80 + j];
      float gir = fmaf(al, er.x, fmaf(be, er.y, er.z));
      float giz = fmaf(al, ez.x, fmaf(be, ez.y, ez.z));
      float gin = fmaf(al, en.x, fmaf(be, en.y, en.z));
      float r = sigm(gir + gh[j] + er.w);
      float z = sigm(giz + gh[40 + j] + ez.w);
      float n = tanh_(fmaf(r, gh[80 + j] + en.w, gin));
      float ho = hf[j];
      float hn = fmaf(z, ho - n, n);      // (1-z)*n + z*h
      hf[j] = hn;
      h16[j] = (_Float16)hn;
    }
    WFENCE();
  }

  // ---------------- switch weights to decoder ----------------
  __syncthreads();
  loadA(A0, dwhh, tid);
  loadA(A1, dwih, tid);
  __syncthreads();

  // ---------------- decoder scan (128 steps, teacher forcing) ----------------
  for (int t = 0; t < LOUT; ++t) {
    if (lane < 16) {
      // dec_x = LayerNorm(relu(x*w + b)) for this row, written as f16 B operand
      float xd = (t == 0) ? in[(size_t)gbrow * LIN + (LIN - 1)]
                          : tgt[(size_t)gbrow * LOUT + (t - 1)];
      float sum = 0.f, ss = 0.f;
#pragma unroll
      for (int j = 0; j < H; ++j) {
        float e = fmaxf(fmaf(xd, DWv[j], DBv[j]), 0.f);
        sum += e; ss += e * e;
      }
      float m = sum * (1.f / H);
      float var = ss * (1.f / H) - m * m;
      float inv = rsqrtf(var + 1e-5f);
      _Float16* dxr = DX + lane * 64;
#pragma unroll
      for (int j = 0; j < H; ++j) {
        float e = fmaxf(fmaf(xd, DWv[j], DBv[j]), 0.f);
        dxr[j] = (_Float16)fmaf((e - m) * inv, LNGv[j], LNBv[j]);
      }
    }
    WFENCE();
    gemm_ldsA(A1, DX,  GIp, lane);   // gi = dec_x @ W_ih^T
    gemm_ldsA(A0, H16, GHp, lane);   // gh = h @ W_hh^T
    WFENCE();
    const float* gh = GHp + rr * 128;
    const float* gi = GIp + rr * 128;
    float* hf = HF + rr * 40;
    _Float16* h16 = H16 + rr * 64;
#pragma unroll 4
    for (int jj = 0; jj < 20; ++jj) {
      int j = j0 + jj;
      float2 pr = D2[j], pz = D2[40 + j], pn = D2[80 + j];
      float r = sigm(gi[j] + pr.x + gh[j] + pr.y);
      float z = sigm(gi[40 + j] + pz.x + gh[40 + j] + pz.y);
      float n = tanh_(fmaf(r, gh[80 + j] + pn.y, gi[80 + j] + pn.x));
      float ho = hf[j];
      float hn = fmaf(z, ho - n, n);
      hf[j] = hn;
      h16[j] = (_Float16)hn;
    }
    WFENCE();
    if (lane < 16) {
      const float* hr = HF + lane * 40;
      float acc = outb;
#pragma unroll
      for (int j = 0; j < H; ++j) acc = fmaf(hr[j], OWv[j], acc);
      out[(size_t)(rowBase + lane) * LOUT + t] = acc;
    }
    WFENCE();
  }
}

extern "C" void kernel_launch(void* const* d_in, const int* in_sizes, int n_in,
                              void* d_out, int out_size, void* d_ws, size_t ws_size,
                              hipStream_t stream) {
  (void)in_sizes; (void)n_in; (void)out_size; (void)d_ws; (void)ws_size;
  const float* p[20];
  for (int i = 0; i < 20; ++i) p[i] = (const float*)d_in[i];
  dim3 grid(Bsz / ROWS);
  dim3 block(WAVES * 32);
  hipLaunchKernelGGL(seq2seq_gru, grid, block, 0, stream,
                     p[0], p[1], p[2], p[3], p[4], p[5], p[6], p[7], p[8], p[9],
                     p[10], p[11], p[12], p[13], p[14], p[15], p[16], p[17], p[18], p[19],
                     (float*)d_out);
}
